// AttentionLayer_79396765434386
// MI455X (gfx1250) — compile-verified, matrix operand order checked by reading, not run on hardware
//
#include <hip/hip_runtime.h>
#include <hip/hip_bf16.h>

// ---------------------------------------------------------------------------
// AttentionLayer for MI455X (gfx1250, wave32, WMMA + TDM).
//  B=16, S=2048, D=1024, d=128.  softmax over the QUERY axis (axis=1).
//   K0: convert W{q,k,v} fp32->bf16 into ws
//   K1: projection GEMM [32768,1024]x[1024,128] -> bf16 Q/K/V (25MB, L2 resident)
//   K2: online column stats (max, sum-exp over q) of scores = Q K^T / sqrt(d);
//       Q tiles staged via TENSOR_LOAD_TO_LDS with pad_enable (bank-conflict-free)
//   K3: H = P V with P = exp(score - colmax)/colsum recomputed on the fly;
//       V tiles staged via TENSOR_LOAD_TO_LDS, P transposed through padded LDS.
// ---------------------------------------------------------------------------

typedef __attribute__((ext_vector_type(16))) __bf16 v16bf;
typedef __attribute__((ext_vector_type(8)))  float  v8f;
typedef __attribute__((ext_vector_type(4)))  unsigned int v4u;
typedef __attribute__((ext_vector_type(8)))  int v8i;
typedef __attribute__((ext_vector_type(4)))  int v4i;

#if __has_builtin(__builtin_amdgcn_tensor_load_to_lds)
#define HAVE_TDM 1
#else
#define HAVE_TDM 0
#endif

__device__ __forceinline__ __bf16 us2bf(unsigned short u) {
  return __builtin_bit_cast(__bf16, u);
}
__device__ __forceinline__ unsigned short f2us_bf(float f) {
  unsigned u = __builtin_bit_cast(unsigned, f);
  unsigned r = u + 0x7FFFu + ((u >> 16) & 1u);   // round-to-nearest-even
  return (unsigned short)(r >> 16);
}

// K-pair base for the 16-bit WMMA A/B operand layout (CDNA5 ISA):
// lane-half 0 owns K {0..7,16..23}, half 1 owns K {8..15,24..31}, 2 K per VGPR.
__device__ __forceinline__ int kpair_base(int j, int half) {
  return (j < 4) ? (half * 8 + 2 * j) : (16 + half * 8 + 2 * (j - 4));
}

__device__ __forceinline__ v8f wmma_bf16(v16bf a, v16bf b, v8f c) {
  return __builtin_amdgcn_wmma_f32_16x16x32_bf16(
      false, a, false, b, (short)0, c, false, false);
}

// Operand gather from a row-major (row = M or N, 32 K columns) LDS tile.
// K pairs are contiguous -> lowers to ds_load_b32/b128, padded ld kills
// the 16-way bank conflict a power-of-two stride would cause.
__device__ __forceinline__ v16bf load_rowmajor_tile(const unsigned short* t,
                                                    int ld, int co) {
  int lane = threadIdx.x & 31, m = lane & 15, half = lane >> 4;
  v16bf a;
#pragma unroll
  for (int j = 0; j < 8; ++j) {
    int kb = kpair_base(j, half);
    a[2 * j]     = us2bf(t[m * ld + co + kb]);
    a[2 * j + 1] = us2bf(t[m * ld + co + kb + 1]);
  }
  return a;
}

// ---------------------------------------------------------------------------
// Tensor Data Mover: 1-D load of `nelem` bf16 elements into LDS.
// D# built per CDNA5 ISA: group0 = {count=1, lds_addr,
// global_addr[56:0], type=2}; group1 = {data_size=2B (+optional LDS padding),
// tensor_dim0=nelem, tensor_dim1=1, tile_dim0=nelem, stride0=nelem}.
// pad flags: pad_enable(bit20) | pad_interval(bits24:22) | pad_amount(bits31:25)
// ---------------------------------------------------------------------------
__device__ __forceinline__ void tdm_load_2b(unsigned lds_off, const void* gptr,
                                            unsigned nelem, unsigned pad_flags) {
#if HAVE_TDM
  unsigned long long ga = (unsigned long long)(size_t)gptr;
  v4u g0 = {1u,                                    // count=1 (user descriptor)
            lds_off,                               // LDS byte address
            (unsigned)ga,                          // global_addr[31:0]
            (unsigned)((ga >> 32) & 0x01FFFFFFu)   // global_addr[56:32]
                | 0x80000000u};                    // type=2 ("image")
  v8i g1 = {(int)(0x00010000u | pad_flags),        // data_size=2B | padding
            (int)(nelem << 16),                    // tensor_dim0[15:0]
            (int)(((nelem >> 16) & 0xFFFFu) | (1u << 16)),  // td0 hi | td1=1
            (int)(nelem << 16),                    // tile_dim0
            0,                                     // tile_dim1/2 unused
            (int)nelem,                            // tensor_dim0_stride lo
            0, 0};
  v4i z4 = {};
  v8i z8 = {};
#if __clang_major__ >= 23
  __builtin_amdgcn_tensor_load_to_lds(g0, g1, z4, z4, z8, 0);
#else
  __builtin_amdgcn_tensor_load_to_lds(g0, g1, z4, z4, 0);
#endif
  __builtin_amdgcn_s_wait_tensorcnt(0);
#endif
}

// ---------------------------------------------------------------------------
// K0: fp32 -> bf16 elementwise convert (weights)
// ---------------------------------------------------------------------------
__global__ void cvt_bf16_kernel(const float* __restrict__ src,
                                unsigned short* __restrict__ dst, int n) {
  int i = blockIdx.x * blockDim.x + threadIdx.x;
  if (i < n) dst[i] = f2us_bf(src[i]);
}

// ---------------------------------------------------------------------------
// K1: projection GEMM  Y[32768,128] = X[32768,1024] * W[1024,128] + bias
//     256 threads (8 waves); M-block = 64 rows; wave w owns N-tile w.
//     B tile staged TRANSPOSED (n-major, ld=34) so operand reads are
//     contiguous ds loads instead of 16 strided ds_load_u16.
// ---------------------------------------------------------------------------
__global__ void __launch_bounds__(256)
proj_kernel(const float* __restrict__ X, const unsigned short* __restrict__ Wbf,
            const float* __restrict__ bias, unsigned short* __restrict__ Ybf) {
  __shared__ unsigned short ldsA[64 * 34];     // padded rows: 4.25 KB
  __shared__ unsigned short ldsBt[128 * 34];   // transposed W tile: 8.5 KB
  const int tid = threadIdx.x, wave = tid >> 5;
  const int lane = tid & 31, n = lane & 15, half = lane >> 4;
  const size_t mblock = blockIdx.x;            // 0..511
  const float* Xblk = X + mblock * 64 * 1024;

  v8f acc[4] = {};
  for (int kk = 0; kk < 1024; kk += 32) {
    // stage A: 64x32 fp32 -> bf16 (padded ld=34)
#pragma unroll
    for (int i = tid; i < 64 * 32; i += 256) {
      int r = i >> 5, c = i & 31;
      ldsA[r * 34 + c] = f2us_bf(Xblk[(size_t)r * 1024 + kk + c]);
    }
    // stage B transposed: W[kk+k][n] -> ldsBt[n*34 + k]
    const unsigned int* Wk = (const unsigned int*)(Wbf + (size_t)kk * 128);
#pragma unroll
    for (int i = tid; i < 32 * 64; i += 256) {
      unsigned int u = Wk[i];
      int k = i >> 6, n2 = (i & 63) << 1;
      ldsBt[n2 * 34 + k] = (unsigned short)(u & 0xFFFFu);
      ldsBt[(n2 + 1) * 34 + k] = (unsigned short)(u >> 16);
    }
    if (kk + 32 < 1024)  // gfx1250 global_prefetch_b8 for next A chunk
      __builtin_prefetch(&Xblk[(size_t)(tid & 63) * 1024 + kk + 32], 0, 3);
    __syncthreads();

    v16bf b = load_rowmajor_tile(ldsBt + (wave * 16) * 34, 34, 0);
#pragma unroll
    for (int mt = 0; mt < 4; ++mt) {
      v16bf a = load_rowmajor_tile(ldsA + mt * 16 * 34, 34, 0);
      acc[mt] = wmma_bf16(a, b, acc[mt]);
    }
    __syncthreads();
  }
  const float bv = bias[wave * 16 + n];
#pragma unroll
  for (int mt = 0; mt < 4; ++mt)
#pragma unroll
    for (int r = 0; r < 8; ++r) {
      size_t row = mblock * 64 + mt * 16 + r + half * 8;
      Ybf[row * 128 + wave * 16 + n] = f2us_bf(acc[mt][r] + bv);
    }
}

// ---------------------------------------------------------------------------
// K2: column softmax stats (max / sum-exp over q axis) of scores.
//     grid = 16*32; 128 threads (4 waves); wave w owns 16 k-columns.
//     Q tile (16x128, contiguous 4KB) staged by TDM with pad_enable:
//     +4 DWORDs every 64 DWORDs -> LDS row stride 136 ushorts (conflict-free).
// ---------------------------------------------------------------------------
__global__ void __launch_bounds__(128)
colstats_kernel(const unsigned short* __restrict__ Qbf,
                const unsigned short* __restrict__ Kbf,
                float* __restrict__ colmax, float* __restrict__ colsum) {
  __shared__ unsigned short ldsQ[16 * 136];    // 4.25 KB (TDM-padded layout)
  const int b = blockIdx.x >> 5, kt = blockIdx.x & 31;
  const int tid = threadIdx.x, wave = tid >> 5;
  const int lane = tid & 31, n = lane & 15, half = lane >> 4;
  const int kc = kt * 64 + wave * 16;
  const float scale = 0.08838834764831845f;    // 1/sqrt(128)
  // pad_enable | pad_interval=64 DWORDs (code 5) | pad_amount=4 DWORDs (code 3)
  const unsigned padQ = (1u << 20) | (5u << 22) | (3u << 25);
  const unsigned ldsQ_off = (unsigned)(size_t)(void*)ldsQ;

  // preload K^T B-operand: column n maps to row (kc+n) of Kbf[b]
  const unsigned short* krow = Kbf + ((size_t)(b * 2048 + kc + n)) * 128;
  v16bf bop[4];
#pragma unroll
  for (int c = 0; c < 4; ++c)
#pragma unroll
    for (int j = 0; j < 8; ++j) {
      int kb = kpair_base(j, half);
      bop[c][2 * j]     = us2bf(krow[c * 32 + kb]);
      bop[c][2 * j + 1] = us2bf(krow[c * 32 + kb + 1]);
    }

  float m = -3.402823466e38f, s = 0.f;
  for (int qt = 0; qt < 128; ++qt) {
    const unsigned short* Qt = Qbf + ((size_t)(b * 2048 + qt * 16)) * 128;
#if HAVE_TDM
    if (wave == 0) tdm_load_2b(ldsQ_off, Qt, 16 * 128, padQ);
#else
#pragma unroll
    for (int i = tid; i < 16 * 128; i += 128)
      ldsQ[(i >> 7) * 136 + (i & 127)] = Qt[i];
#endif
    __syncthreads();

    v8f sc = {};
#pragma unroll
    for (int c = 0; c < 4; ++c) {
      v16bf a = load_rowmajor_tile(ldsQ, 136, c * 32);
      sc = wmma_bf16(a, bop[c], sc);
    }
    __syncthreads();

    float tm = -3.402823466e38f;
#pragma unroll
    for (int r = 0; r < 8; ++r) tm = fmaxf(tm, sc[r] * scale);
    tm = fmaxf(tm, __shfl_xor(tm, 16));        // combine both lane halves
    float mn = fmaxf(m, tm);
    float ts = 0.f;
#pragma unroll
    for (int r = 0; r < 8; ++r) ts += __expf(sc[r] * scale - mn);
    ts += __shfl_xor(ts, 16);
    s = s * __expf(m - mn) + ts;
    m = mn;
  }
  if (half == 0) {
    colmax[b * 2048 + kc + n] = m;
    colsum[b * 2048 + kc + n] = s;
  }
}

// ---------------------------------------------------------------------------
// K3: H = P V, P computed on the fly from recomputed scores + K2 stats.
//     grid = 16*128 q-tiles; 256 threads (8 waves); wave w owns e-tile w.
//     Waves 0,1 rebuild the 16x32 probability tile per k-chunk (LDS transpose,
//     padded ld=34); wave 7 TDM-stages the 32x128 V tile (8KB contiguous).
// ---------------------------------------------------------------------------
__global__ void __launch_bounds__(256)
hgemm_kernel(const unsigned short* __restrict__ Qbf,
             const unsigned short* __restrict__ Kbf,
             const unsigned short* __restrict__ Vbf,
             const float* __restrict__ colmax, const float* __restrict__ colsum,
             float* __restrict__ out) {
  __shared__ unsigned short ldsP[16 * 34];     // probability tile (padded)
  __shared__ unsigned short ldsV[32 * 128];    // raw V tile (TDM target), 8 KB
  const int b = blockIdx.x >> 7, qt = blockIdx.x & 127;
  const int tid = threadIdx.x, wave = tid >> 5;
  const int lane = tid & 31, n = lane & 15, half = lane >> 4;
  const float scale = 0.08838834764831845f;
  const unsigned ldsV_off = (unsigned)(size_t)(void*)ldsV;

  // waves 0,1: preload Q A-operand for this q-tile (row m = lane&15)
  v16bf aq[4] = {};
  if (wave < 2) {
    const unsigned short* qrow = Qbf + ((size_t)(b * 2048 + qt * 16 + n)) * 128;
#pragma unroll
    for (int c = 0; c < 4; ++c)
#pragma unroll
      for (int j = 0; j < 8; ++j) {
        int kb = kpair_base(j, half);
        aq[c][2 * j]     = us2bf(qrow[c * 32 + kb]);
        aq[c][2 * j + 1] = us2bf(qrow[c * 32 + kb + 1]);
      }
  }

  v8f acc = {};
  const int ecol = wave * 16 + n;
  const unsigned short* Vbat = Vbf + ((size_t)b * 2048) * 128;

  for (int kc = 0; kc < 2048; kc += 32) {
    const unsigned short* vt = Vbat + (size_t)kc * 128;
#if HAVE_TDM
    if (wave == 7) tdm_load_2b(ldsV_off, vt, 32 * 128, 0);
#else
#pragma unroll
    for (int i = tid; i < 32 * 128; i += 256) ldsV[i] = vt[i];
#endif
    if (wave < 2) {
      const int kb0 = kc + wave * 16;
      const unsigned short* krow = Kbf + ((size_t)(b * 2048 + kb0 + n)) * 128;
      v8f sc = {};
#pragma unroll
      for (int c = 0; c < 4; ++c) {
        v16bf bk;
#pragma unroll
        for (int j = 0; j < 8; ++j) {
          int kb = kpair_base(j, half);
          bk[2 * j]     = us2bf(krow[c * 32 + kb]);
          bk[2 * j + 1] = us2bf(krow[c * 32 + kb + 1]);
        }
        sc = wmma_bf16(aq[c], bk, sc);
      }
      const float mcol = colmax[b * 2048 + kb0 + n];
      const float rinv = 1.f / colsum[b * 2048 + kb0 + n];
#pragma unroll
      for (int r = 0; r < 8; ++r) {
        float p = __expf(sc[r] * scale - mcol) * rinv;
        ldsP[(r + half * 8) * 34 + wave * 16 + n] = f2us_bf(p);
      }
    }
    __syncthreads();

    v16bf ap = load_rowmajor_tile(ldsP, 34, 0);   // transposed prob tile
    v16bf bv;                                     // V operand from staged LDS
#pragma unroll
    for (int j = 0; j < 8; ++j) {
      int kb = kpair_base(j, half);
      bv[2 * j]     = us2bf(ldsV[kb * 128 + ecol]);
      bv[2 * j + 1] = us2bf(ldsV[(kb + 1) * 128 + ecol]);
    }
    acc = wmma_bf16(ap, bv, acc);
    __syncthreads();
  }

  float* orow = out + ((size_t)(b * 2048 + qt * 16)) * 128 + ecol;
#pragma unroll
  for (int r = 0; r < 8; ++r)
    orow[(size_t)(r + half * 8) * 128] = acc[r];
}

// ---------------------------------------------------------------------------
extern "C" void kernel_launch(void* const* d_in, const int* in_sizes, int n_in,
                              void* d_out, int out_size, void* d_ws, size_t ws_size,
                              hipStream_t stream) {
  const float* inp_q = (const float*)d_in[0];
  const float* inp_k = (const float*)d_in[1];
  const float* inp_v = (const float*)d_in[2];
  const float* Wq = (const float*)d_in[3];
  const float* bq = (const float*)d_in[4];
  const float* Wk = (const float*)d_in[5];
  const float* bk = (const float*)d_in[6];
  const float* Wv = (const float*)d_in[7];
  const float* bv = (const float*)d_in[8];

  char* ws = (char*)d_ws;
  unsigned short* Qbf = (unsigned short*)(ws);                  // 8.39 MB
  unsigned short* Kbf = (unsigned short*)(ws + 8388608);        // 8.39 MB
  unsigned short* Vbf = (unsigned short*)(ws + 16777216);       // 8.39 MB
  unsigned short* Wqb = (unsigned short*)(ws + 25165824);       // 256 KB
  unsigned short* Wkb = (unsigned short*)(ws + 25427968);
  unsigned short* Wvb = (unsigned short*)(ws + 25690112);
  float* colmax = (float*)(ws + 25952256);                      // 128 KB
  float* colsum = (float*)(ws + 26083328);                      // 128 KB

  cvt_bf16_kernel<<<512, 256, 0, stream>>>(Wq, Wqb, 131072);
  cvt_bf16_kernel<<<512, 256, 0, stream>>>(Wk, Wkb, 131072);
  cvt_bf16_kernel<<<512, 256, 0, stream>>>(Wv, Wvb, 131072);

  proj_kernel<<<512, 256, 0, stream>>>(inp_q, Wqb, bq, Qbf);
  proj_kernel<<<512, 256, 0, stream>>>(inp_k, Wkb, bk, Kbf);
  proj_kernel<<<512, 256, 0, stream>>>(inp_v, Wvb, bv, Vbf);

  colstats_kernel<<<512, 128, 0, stream>>>(Qbf, Kbf, colmax, colsum);

  hgemm_kernel<<<2048, 256, 0, stream>>>(Qbf, Kbf, Vbf, colmax, colsum,
                                         (float*)d_out);
}
